// DecoderLSTM_8641474200440
// MI455X (gfx1250) — matrix operand order repaired and checked by reading, not compiled, not run on hardware
//
#include <hip/hip_runtime.h>
#include <hip/hip_bf16.h>

typedef __attribute__((ext_vector_type(16))) __bf16 v16bf;
typedef __attribute__((ext_vector_type(8)))  __bf16 v8bf;
typedef __attribute__((ext_vector_type(8)))  float  v8f;
typedef __attribute__((ext_vector_type(4)))  float  v4f;
typedef __attribute__((ext_vector_type(4)))  int    v4i;

#define B_   32
#define F_   128
#define N_   1024
#define S_   12
#define H_   256
#define P_   24
#define G4   1024            // 4*H
#define BN_  (B_*N_)         // 32768
#define ROWS 32              // BN rows per workgroup

// ---- workspace layout (bytes) ----
#define WS_WIH   0           // 1024x128 bf16 = 262144
#define WS_WHH   262144      // 1024x256 bf16 = 524288
#define WS_BIAS  786432      // 1024 f32     =   4096

// ---- LDS layout (bytes) ----
#define OFF_GX    0          // 128 KB : 8 waves x 16KB gx fragment dump
#define OFF_UNION 131072     // 128 KB : W chunk double-buffer (2x64KB) / 4 gate bufs (32KB each)
#define OFF_C     262144     //  32 KB : c state, fragment layout
#define OFF_H     294912     //  16 KB : h bf16 [32][256]; x_last bf16 [32][128] in prologue
#define OFF_CW    311296     //   1 KB : conv_w
#define OFF_Y     312320     // 128 B  : per-row y partials
#define LDS_BYTES 312448

// ---- CDNA5 async global->LDS path (guarded; falls back to sync copy) ----
#if defined(__has_builtin)
#  if __has_builtin(__builtin_amdgcn_global_load_async_to_lds_b128)
#    define HAVE_ASYNC_LDS 1
#  endif
#endif
#ifndef HAVE_ASYNC_LDS
#  define HAVE_ASYNC_LDS 0
#endif

#if HAVE_ASYNC_LDS
typedef __attribute__((address_space(1))) v4i gv4i;   // global int4
typedef __attribute__((address_space(3))) v4i lv4i;   // LDS int4
#  if defined(__has_builtin) && __has_builtin(__builtin_amdgcn_s_wait_asynccnt)
#    define ASYNC_WAIT0() __builtin_amdgcn_s_wait_asynccnt(0)
#  else
#    define ASYNC_WAIT0() asm volatile("s_wait_asynccnt 0" ::: "memory")
#  endif
#else
#  define ASYNC_WAIT0() ((void)0)
#endif

__global__ void __launch_bounds__(256)
lstm_prep_kernel(const float* __restrict__ W_ih, const float* __restrict__ W_hh,
                 const float* __restrict__ b_ih, const float* __restrict__ b_hh,
                 char* __restrict__ ws)
{
    int idx = blockIdx.x * blockDim.x + threadIdx.x;
    __bf16* wih  = (__bf16*)(ws + WS_WIH);
    __bf16* whh  = (__bf16*)(ws + WS_WHH);
    float*  bias = (float*)(ws + WS_BIAS);
    if (idx < G4 * F_) wih[idx] = (__bf16)W_ih[idx];
    if (idx < G4 * H_) whh[idx] = (__bf16)W_hh[idx];
    if (idx < G4)      bias[idx] = b_ih[idx] + b_hh[idx];
}

__device__ __forceinline__ float fast_sigmoid(float x) {
    return 1.0f / (1.0f + __expf(-x));
}
__device__ __forceinline__ float fast_tanh(float x) {
    float e = __expf(2.0f * x);
    return (e - 1.0f) / (e + 1.0f);
}

// Stage W[:, k0:k0+32] (1024 rows) into LDS buffer as [j][32] bf16 (64 KB).
// Async path: per-lane GLOBAL_LOAD_ASYNC_TO_LDS_B128 (ASYNCcnt-tracked, no VGPR data).
__device__ __forceinline__ void stage_chunk(char* smem, int bufOff,
                                            const char* wsrc, int rowBytes,
                                            int k0, int tid)
{
#if HAVE_ASYNC_LDS
#pragma unroll
    for (int i = 0; i < 16; ++i) {
        int idx = tid + 256 * i;        // 0..4095 16B transfers
        int j = idx >> 2, part = idx & 3;
        gv4i* g = (gv4i*)(wsrc + (size_t)j * rowBytes + k0 * 2 + part * 16);
        lv4i* l = (lv4i*)(smem + bufOff + j * 64 + part * 16);
        __builtin_amdgcn_global_load_async_to_lds_b128(g, l, 0, 0);
    }
#else
#pragma unroll
    for (int i = 0; i < 16; ++i) {
        int idx = tid + 256 * i;
        int j = idx >> 2, part = idx & 3;
        *(v4f*)(smem + bufOff + j * 64 + part * 16) =
            *(const v4f*)(wsrc + (size_t)j * rowBytes + k0 * 2 + part * 16);
    }
#endif
}

__global__ void __launch_bounds__(256)
lstm_main_kernel(const float* __restrict__ x,
                 const float* __restrict__ conv_w,
                 const float* __restrict__ conv_b,
                 const char*  __restrict__ ws,
                 float*       __restrict__ out)
{
    extern __shared__ char smem[];
    const int tid  = threadIdx.x;
    const int lane = tid & 31;
    const int wv   = tid >> 5;     // 0..7
    const int m    = wv & 1;       // row half of the 32-row tile
    const int ng   = wv >> 1;      // 256-col group == gate index
    const int l15  = lane & 15;
    const int lhi  = lane >> 4;    // 0/1
    const int bn0  = blockIdx.x * ROWS;

    const char* wih_b = ws + WS_WIH;
    const char* whh_b = ws + WS_WHH;
    const float* bias = (const float*)(ws + WS_BIAS);

    // ---------------- prologue ----------------
    if (tid < H_) ((float*)(smem + OFF_CW))[tid] = conv_w[tid];
    {   // zero c state
        float* c = (float*)(smem + OFF_C);
        for (int i = tid; i < ROWS * H_; i += 256) c[i] = 0.0f;
    }
    {   // gather x_last -> bf16 [32][128] at OFF_H
        __bf16* xb = (__bf16*)(smem + OFF_H);
#pragma unroll
        for (int i = 0; i < 16; ++i) {
            int e    = tid + 256 * i;      // 0..4095
            int row  = e >> 7;
            int fp   = e & 127;
            int bn   = bn0 + row;
            int b    = bn >> 10;
            int n    = bn & 1023;
            int q    = 1408 + fp;
            int fidx = q / 12;
            int sidx = q - fidx * 12;
            float v  = x[(size_t)b * (F_ * N_ * S_) + (size_t)fidx * (N_ * S_) + n * S_ + sidx];
            xb[row * F_ + fp] = (__bf16)v;
        }
    }

    v8f acc[16];
#pragma unroll
    for (int t = 0; t < 16; ++t) acc[t] = (v8f){0,0,0,0,0,0,0,0};

    // gx = x_last @ W_ih^T  (K=128 -> 4 chunks of 32, double-buffered)
    stage_chunk(smem, OFF_UNION, wih_b, F_ * 2, 0, tid);
    ASYNC_WAIT0();
    __syncthreads();
    for (int kc = 0; kc < 4; ++kc) {
        const int k0 = kc * 32;
        const int buf = OFF_UNION + (kc & 1) * 65536;
        if (kc < 3)   // prefetch next chunk into the other buffer
            stage_chunk(smem, OFF_UNION + ((kc + 1) & 1) * 65536, wih_b, F_ * 2, k0 + 32, tid);
        v16bf Afrag;
        {   // A: lanes 0-15 hold K {0..15}, lanes 16-31 hold {16..31}
            const char* xb = smem + OFF_H;
            int row = m * 16 + l15;
            int o1  = row * (F_ * 2) + (k0 + lhi * 16) * 2;
            v8bf a0 = *(const v8bf*)(xb + o1);
            v8bf a1 = *(const v8bf*)(xb + o1 + 16);
            Afrag = __builtin_shufflevector(a0, a1, 0,1,2,3,4,5,6,7,8,9,10,11,12,13,14,15);
        }
#pragma unroll
        for (int t = 0; t < 16; ++t) {
            int jj = ng * 256 + t * 16 + l15;   // B: lane n=l15, K half by lhi
            const v16bf Bfrag = *(const v16bf*)(smem + buf + jj * 64 + lhi * 32);
            acc[t] = __builtin_amdgcn_wmma_f32_16x16x32_bf16(
                false, Afrag, false, Bfrag, (short)0, acc[t], false, false);
        }
        if (kc < 3) ASYNC_WAIT0();
        __syncthreads();
    }
    {   // add fused bias, dump gx fragments to LDS
        char* gx = smem + OFF_GX + wv * 16384;
#pragma unroll
        for (int t = 0; t < 16; ++t) {
            float bv = bias[ng * 256 + t * 16 + l15];
#pragma unroll
            for (int r = 0; r < 8; ++r) acc[t][r] += bv;
            *(v8f*)(gx + t * 1024 + lane * 32) = acc[t];
        }
    }
    __syncthreads();

    // ---------------- recurrence ----------------
    const float cb = conv_b[0];
    for (int p = 0; p < P_; ++p) {
        {   // acc = gx
            const char* gx = smem + OFF_GX + wv * 16384;
#pragma unroll
            for (int t = 0; t < 16; ++t)
                acc[t] = *(const v8f*)(gx + t * 1024 + lane * 32);
        }
        if (p > 0) {
            // K=256 -> 8 chunks of 32, async double-buffered
            stage_chunk(smem, OFF_UNION, whh_b, H_ * 2, 0, tid);
            ASYNC_WAIT0();
            __syncthreads();
            for (int kc = 0; kc < 8; ++kc) {
                const int k0 = kc * 32;
                const int buf = OFF_UNION + (kc & 1) * 65536;
                if (kc < 7)
                    stage_chunk(smem, OFF_UNION + ((kc + 1) & 1) * 65536, whh_b, H_ * 2, k0 + 32, tid);
                v16bf Afrag;
                {
                    const char* hb = smem + OFF_H;
                    int row = m * 16 + l15;
                    int o1  = row * (H_ * 2) + (k0 + lhi * 16) * 2;
                    v8bf a0 = *(const v8bf*)(hb + o1);
                    v8bf a1 = *(const v8bf*)(hb + o1 + 16);
                    Afrag = __builtin_shufflevector(a0, a1, 0,1,2,3,4,5,6,7,8,9,10,11,12,13,14,15);
                }
#pragma unroll
                for (int t = 0; t < 16; ++t) {
                    int jj = ng * 256 + t * 16 + l15;
                    const v16bf Bfrag = *(const v16bf*)(smem + buf + jj * 64 + lhi * 32);
                    acc[t] = __builtin_amdgcn_wmma_f32_16x16x32_bf16(
                        false, Afrag, false, Bfrag, (short)0, acc[t], false, false);
                }
                if (kc < 7) ASYNC_WAIT0();
                __syncthreads();
            }
        } else {
            __syncthreads();   // keep gate-region producers/consumers separated at p==0
        }
        // union region free to hold gates
        if (tid < ROWS) ((float*)(smem + OFF_Y))[tid] = 0.0f;
        {   // activate own gate slice: ng 0:i 1:f -> sigmoid, 2:g -> tanh, 3:o -> sigmoid
            char* gbuf = smem + OFF_UNION + ng * 32768 + m * 16384;
#pragma unroll
            for (int t = 0; t < 16; ++t) {
                v8f v = acc[t];
#pragma unroll
                for (int r = 0; r < 8; ++r)
                    v[r] = (ng == 2) ? fast_tanh(v[r]) : fast_sigmoid(v[r]);
                *(v8f*)(gbuf + t * 1024 + lane * 32) = v;
            }
        }
        __syncthreads();
        {   // elementwise LSTM update in fragment space: thread = (lane l, reg r)
            const float* iG = (const float*)(smem + OFF_UNION + 0 * 32768);
            const float* fG = (const float*)(smem + OFF_UNION + 1 * 32768);
            const float* gG = (const float*)(smem + OFF_UNION + 2 * 32768);
            const float* oG = (const float*)(smem + OFF_UNION + 3 * 32768);
            float*  cS = (float*)(smem + OFF_C);
            __bf16* hb = (__bf16*)(smem + OFF_H);
            const float* cw = (const float*)(smem + OFF_CW);
            float* yl = (float*)(smem + OFF_Y);
            int l = tid & 31, r = tid >> 5;
            int row_base = r + ((l >> 4) << 3);
            int col_lo   = l & 15;
#pragma unroll
            for (int mm = 0; mm < 2; ++mm) {
                int row = mm * 16 + row_base;
                float yacc = 0.0f;
#pragma unroll
                for (int t = 0; t < 16; ++t) {
                    int off = (mm * 16384 + t * 1024 + l * 32 + r * 4) >> 2;  // float idx
                    float cnew = fG[off] * cS[off] + iG[off] * gG[off];
                    cS[off] = cnew;
                    float h = oG[off] * fast_tanh(cnew);
                    int col = t * 16 + col_lo;
                    hb[row * H_ + col] = (__bf16)h;
                    yacc += h * cw[col];
                }
                atomicAdd(yl + row, yacc);
            }
        }
        __syncthreads();
        if (tid < ROWS) {
            float yv = ((const float*)(smem + OFF_Y))[tid] + cb;
            size_t bn = (size_t)(bn0 + tid);
            out[bn * P_ + p] = yv;
            out[(size_t)BN_ * P_ + bn * P_ + p] = yv;   // duplicated tuple output
        }
    }
}

extern "C" void kernel_launch(void* const* d_in, const int* in_sizes, int n_in,
                              void* d_out, int out_size, void* d_ws, size_t ws_size,
                              hipStream_t stream)
{
    const float* x     = (const float*)d_in[0];
    const float* W_ih  = (const float*)d_in[1];
    const float* W_hh  = (const float*)d_in[2];
    const float* b_ih  = (const float*)d_in[3];
    const float* b_hh  = (const float*)d_in[4];
    const float* convw = (const float*)d_in[5];
    const float* convb = (const float*)d_in[6];
    char* ws = (char*)d_ws;

    // convert weights to bf16, fuse biases (262144 elements max -> 1024 blocks x 256)
    lstm_prep_kernel<<<1024, 256, 0, stream>>>(W_ih, W_hh, b_ih, b_hh, ws);
    // one workgroup per 32 BN rows
    lstm_main_kernel<<<BN_ / ROWS, 256, LDS_BYTES, stream>>>(x, convw, convb, ws, (float*)d_out);
}
